// Model_64261300683328
// MI455X (gfx1250) — matrix operand order repaired and checked by reading, not compiled, not run on hardware
//
#include <hip/hip_runtime.h>
#include <hip/hip_bf16.h>
#include <math.h>

// ---------------------------------------------------------------------------
// MI455X / gfx1250 conv-autoencoder forward pass.
// - intermediate activations: f16 NHWC (channel-contiguous => vector loads)
// - conv/fc matmuls: v_wmma_f32_16x16x32_f16, f32 accumulate, NT-wide
//   output-channel register tiling (independent WMMAs hide D->A/B hazards)
// - conv weights packed f16 [tap][CoutP][CinP], staged into LDS via
//   global_load_async_to_lds_b128 (ASYNCcnt) and read as 32B ds_load fragments
// ---------------------------------------------------------------------------

#define USE_ASYNC_LDS 1

typedef __attribute__((ext_vector_type(16))) _Float16 v16h;
typedef __attribute__((ext_vector_type(8)))  _Float16 v8h;
typedef __attribute__((ext_vector_type(8)))  float    v8f;

// WMMA fragment maps used here (A/B: this code's contiguous k order, not the ISA table; C/D: ISA):
//  A (16x32 MxK):  m = lane&15; elem i -> k = i + 16*(lane>>4)
//  B (32x16 KxN):  n = lane&15; elem i -> k = i + 16*(lane>>4)
//  C/D (16x16):    vgpr r -> M = r + 8*(lane>>4), N = lane&15

// ---------------------------------------------------------------------------
// 3x3 SAME conv, implicit GEMM, NHWC f16.  Block = 4 waves = 4 pixel tiles.
// Each wave: 16 pixels x (16*NT) out-channels.
// ---------------------------------------------------------------------------
template<int CIN, int COUT, int H, int W, int ACT, bool OUTF32, int NT>
__global__ __launch_bounds__(128)
void conv3x3_wmma(const _Float16* __restrict__ in, const _Float16* __restrict__ wt,
                  const float* __restrict__ bias, void* __restrict__ outv)
{
  constexpr int CinP  = (CIN  + 31) & ~31;
  constexpr int CoutP = (COUT + 15) & ~15;
  constexpr int HW    = H * W;
  __shared__ __align__(16) _Float16 lds_w[9 * 16 * NT * CinP];

  const int tid  = threadIdx.x;
  const int lane = tid & 31;
  const int wv   = tid >> 5;
  const int hw   = lane >> 4;
  const int co0  = blockIdx.y * (16 * NT);
  const int b    = blockIdx.z;
  const int m0   = blockIdx.x * 64 + wv * 16;

  { // stage this block's (16*NT)-wide Cout slice of packed weights into LDS
    constexpr int SLICE = 16 * NT * CinP * 2;         // bytes per tap
    constexpr int TOTB  = 9 * SLICE;
#if USE_ASYNC_LDS
    for (int e = tid; e * 16 < TOTB; e += 128) {
      const int byte = e * 16;
      const int t = byte / SLICE;
      const int r = byte % SLICE;
      const char* g = reinterpret_cast<const char*>(
          &wt[((size_t)t * CoutP + co0) * CinP]) + r;
      unsigned lds_addr = (unsigned)(size_t)((char*)lds_w + byte);
      asm volatile("global_load_async_to_lds_b128 %0, %1, off"
                   :: "v"(lds_addr), "v"(g) : "memory");
    }
    asm volatile("s_wait_asynccnt 0x0" ::: "memory");
#else
    uint4* d = reinterpret_cast<uint4*>(lds_w);
    for (int e = tid; e * 16 < TOTB; e += 128) {
      const int byte = e * 16;
      const int t = byte / SLICE;
      const int r = byte % SLICE;
      d[e] = *reinterpret_cast<const uint4*>(
          reinterpret_cast<const char*>(&wt[((size_t)t * CoutP + co0) * CinP]) + r);
    }
#endif
  }
  __syncthreads();

  const _Float16* inb = in + (size_t)b * HW * CIN;
  const int p  = m0 + (lane & 15);
  const int py = p / W, px = p % W;

  v8f acc[NT];
  #pragma unroll
  for (int j = 0; j < NT; ++j) acc[j] = (v8f){};

  #pragma unroll
  for (int t = 0; t < 9; ++t) {
    const int  y   = py + t / 3 - 1;
    const int  x   = px + t % 3 - 1;
    const bool vld = (y >= 0) && (y < H) && (x >= 0) && (x < W);
    const int  off = (min(max(y, 0), H - 1) * W + min(max(x, 0), W - 1));
    #pragma unroll
    for (int c0 = 0; c0 < CinP; c0 += 32) {
      v16h av;
      if constexpr (CIN % 8 == 0) {
        const _Float16* pp = inb + (size_t)off * CIN + c0 + 16 * hw;
        v8h lo = *reinterpret_cast<const v8h*>(pp);
        v8h hi = *reinterpret_cast<const v8h*>(pp + 8);
        #pragma unroll
        for (int i = 0; i < 8; ++i) {
          av[i]     = vld ? lo[i] : (_Float16)0.f;
          av[i + 8] = vld ? hi[i] : (_Float16)0.f;
        }
      } else {
        #pragma unroll
        for (int i = 0; i < 16; ++i) {
          const int  k  = i + 16 * hw;
          const int  c  = c0 + k;
          const bool ok = vld && (c < CIN);
          _Float16 v = inb[(size_t)off * CIN + (c < CIN ? c : CIN - 1)];
          av[i] = ok ? v : (_Float16)0.f;
        }
      }
      #pragma unroll
      for (int j = 0; j < NT; ++j) {
        const v16h bv = *reinterpret_cast<const v16h*>(
            &lds_w[((t * 16 * NT) + j * 16 + (lane & 15)) * CinP + c0 + 16 * hw]);
        acc[j] = __builtin_amdgcn_wmma_f32_16x16x32_f16(
                   false, av, false, bv, (short)0, acc[j], false, false);
      }
    }
  }

  const int n  = lane & 15;
  const int mb = hw * 8;
  #pragma unroll
  for (int j = 0; j < NT; ++j) {
    const int co = co0 + j * 16 + n;
    if (co < COUT) {
      const float bvv = bias[co];
      float vs[8];
      #pragma unroll
      for (int r = 0; r < 8; ++r) {
        float v = acc[j][r] + bvv;
        if (ACT == 1) v = fmaxf(v, 0.0f);
        if (ACT == 2) v = 1.0f / (1.0f + __expf(-v));
        vs[r] = v;
      }
      if constexpr (OUTF32) {
        float* ob = (float*)outv + ((size_t)b * COUT + co) * HW + m0 + mb;
        float4 lo = {vs[0], vs[1], vs[2], vs[3]};
        float4 hi = {vs[4], vs[5], vs[6], vs[7]};
        *reinterpret_cast<float4*>(ob)     = lo;
        *reinterpret_cast<float4*>(ob + 4) = hi;
      } else {
        _Float16* ob = (_Float16*)outv + ((size_t)b * HW + m0 + mb) * COUT + co;
        #pragma unroll
        for (int r = 0; r < 8; ++r)
          ob[(size_t)r * COUT] = (_Float16)vs[r];
      }
    }
  }
}

// ---------------------------------------------------------------------------
// WMMA GEMM: out[M,N] = act(A[M,K] @ BT^T + bias).  BT: f16 [N][K].
// Wave computes 16 x (16*NT).  PERMB: decoder-fc bias column permute.
// ---------------------------------------------------------------------------
template<int ACT, bool PERMB, int NT>
__global__ __launch_bounds__(32)
void gemm_wmma(const _Float16* __restrict__ A, const _Float16* __restrict__ BT,
               const float* __restrict__ bias, _Float16* __restrict__ out,
               int N, int K)
{
  const int lane = threadIdx.x & 31;
  const int hw   = lane >> 4;
  const int m0   = blockIdx.x * 16;
  const int n0   = blockIdx.y * (16 * NT);

  const _Float16* ap = &A[(size_t)(m0 + (lane & 15)) * K + 16 * hw];
  const _Float16* bp[NT];
  #pragma unroll
  for (int j = 0; j < NT; ++j)
    bp[j] = &BT[(size_t)(n0 + j * 16 + (lane & 15)) * K + 16 * hw];

  v8f acc[NT];
  #pragma unroll
  for (int j = 0; j < NT; ++j) acc[j] = (v8f){};

  for (int k0 = 0; k0 < K; k0 += 32) {
    if (k0 + 32 < K) {
      __builtin_prefetch(ap + k0 + 32, 0, 1);
      __builtin_prefetch(bp[0] + k0 + 32, 0, 1);
    }
    v8h lo = *reinterpret_cast<const v8h*>(ap + k0);
    v8h hi = *reinterpret_cast<const v8h*>(ap + k0 + 8);
    v16h av;
    #pragma unroll
    for (int i = 0; i < 8; ++i) { av[i] = lo[i]; av[i + 8] = hi[i]; }
    #pragma unroll
    for (int j = 0; j < NT; ++j) {
      const v16h bv = *reinterpret_cast<const v16h*>(bp[j] + k0);
      acc[j] = __builtin_amdgcn_wmma_f32_16x16x32_f16(
                 false, av, false, bv, (short)0, acc[j], false, false);
    }
  }

  const int mb = m0 + hw * 8;
  #pragma unroll
  for (int j = 0; j < NT; ++j) {
    const int n  = n0 + j * 16 + (lane & 15);
    const int bi = PERMB ? ((n & 63) * 64 + (n >> 6)) : n;
    const float bvv = bias[bi];
    #pragma unroll
    for (int r = 0; r < 8; ++r) {
      float v = acc[j][r] + bvv;
      if (ACT == 1) v = fmaxf(v, 0.0f);
      out[(size_t)(mb + r) * N + n] = (_Float16)v;
    }
  }
}

// ---------------------------------------------------------------------------
// Packing kernels
// ---------------------------------------------------------------------------
__global__ void pack_conv_w(const float* __restrict__ src, _Float16* __restrict__ dst,
                            int COUT, int CIN, int CoutP, int CinP)
{
  int i = blockIdx.x * blockDim.x + threadIdx.x;
  int tot = 9 * CoutP * CinP;
  if (i >= tot) return;
  int c  = i % CinP;
  int co = (i / CinP) % CoutP;
  int t  = i / (CinP * CoutP);
  float v = (co < COUT && c < CIN) ? src[(size_t)(co * CIN + c) * 9 + t] : 0.0f;
  dst[i] = (_Float16)v;
}

__global__ void pack_fc1_w(const float* __restrict__ src, _Float16* __restrict__ dst)
{ // src [K=4096 (NCHW k=c*64+s)][N=1024] -> dst [N][K (NHWC k=s*64+c)]
  int i = blockIdx.x * blockDim.x + threadIdx.x;
  if (i >= 1024 * 4096) return;
  int k = i & 4095, n = i >> 12;
  int s = k >> 6,   c = k & 63;
  dst[i] = (_Float16)src[(size_t)(c * 64 + s) * 1024 + n];
}

__global__ void pack_dfc_w(const float* __restrict__ src, _Float16* __restrict__ dst)
{ // src [K=128][N=4096 (NCHW col c*64+s)] -> dst [N (NHWC n=s*64+c)][K]
  int i = blockIdx.x * blockDim.x + threadIdx.x;
  if (i >= 4096 * 128) return;
  int k = i & 127, n = i >> 7;
  int s = n >> 6,  c = n & 63;
  dst[i] = (_Float16)src[(size_t)k * 4096 + (c * 64 + s)];
}

__global__ void pack_x(const float* __restrict__ src, _Float16* __restrict__ dst)
{ // x NCHW f32 -> NHWC f16
  int i = blockIdx.x * blockDim.x + threadIdx.x;
  if (i >= 128 * 4096 * 3) return;
  int c = i % 3, p = (i / 3) & 4095, b = i / (3 * 4096);
  dst[i] = (_Float16)src[((size_t)b * 3 + c) * 4096 + p];
}

// ---------------------------------------------------------------------------
__global__ void fc2_kernel(const _Float16* __restrict__ h, const float* __restrict__ w,
                           const float* __restrict__ bias, float* __restrict__ params)
{
  int tid = blockIdx.x * blockDim.x + threadIdx.x;
  if (tid >= 128 * 3) return;
  int b = tid / 3, j = tid % 3;
  float acc = bias[j];
  for (int k = 0; k < 1024; ++k)
    acc += (float)h[(size_t)b * 1024 + k] * w[k * 3 + j];
  params[tid] = acc;
}

__global__ void maxpool2_nhwc(const _Float16* __restrict__ in, _Float16* __restrict__ out,
                              int Bn, int H, int W, int C)
{
  int Ho = H >> 1, Wo = W >> 1;
  int total = Bn * Ho * Wo * C;
  int i = blockIdx.x * blockDim.x + threadIdx.x;
  if (i >= total) return;
  int c = i % C, x = (i / C) % Wo, y = (i / (C * Wo)) % Ho, b = i / (C * Wo * Ho);
  const _Float16* p = in + (((size_t)b * H + 2 * y) * W + 2 * x) * C + c;
  float v = fmaxf(fmaxf((float)p[0], (float)p[C]),
                  fmaxf((float)p[(size_t)W * C], (float)p[(size_t)W * C + C]));
  out[i] = (_Float16)v;
}

__global__ void up2_nhwc(const _Float16* __restrict__ in, _Float16* __restrict__ out,
                         int Bn, int H, int W, int C)
{
  int Ho = 2 * H, Wo = 2 * W;
  int total = Bn * Ho * Wo * C;
  int i = blockIdx.x * blockDim.x + threadIdx.x;
  if (i >= total) return;
  int c = i % C, x = (i / C) % Wo, y = (i / (C * Wo)) % Ho, b = i / (C * Wo * Ho);
  float sy = (y + 0.5f) * 0.5f - 0.5f;
  float sx = (x + 0.5f) * 0.5f - 0.5f;
  int y0 = (int)floorf(sy), x0 = (int)floorf(sx);
  float fy = sy - (float)y0, fx = sx - (float)x0;
  int y0c = min(max(y0, 0), H - 1), y1c = min(max(y0 + 1, 0), H - 1);
  int x0c = min(max(x0, 0), W - 1), x1c = min(max(x0 + 1, 0), W - 1);
  const _Float16* p = in + (size_t)b * H * W * C + c;
  float v00 = (float)p[((size_t)y0c * W + x0c) * C];
  float v01 = (float)p[((size_t)y0c * W + x1c) * C];
  float v10 = (float)p[((size_t)y1c * W + x0c) * C];
  float v11 = (float)p[((size_t)y1c * W + x1c) * C];
  out[i] = (_Float16)((1.f - fy) * ((1.f - fx) * v00 + fx * v01)
                    +        fy  * ((1.f - fx) * v10 + fx * v11));
}

__global__ void sample_gather_kernel(const float* __restrict__ params,
                                     const float* __restrict__ u_glob,
                                     const float* __restrict__ u_loc,
                                     const float* __restrict__ latent,
                                     _Float16* __restrict__ code)
{
  __shared__ float s_w[10];
  __shared__ int   s_idx[10];
  const int b = blockIdx.x;

  if (threadIdx.x == 0) {
    const float S = 32.0f, E = 1e-6f;
    float mean0 = fminf(fmaxf(params[b * 3 + 0], 0.f), S - 1.f);
    float mean1 = fminf(fmaxf(params[b * 3 + 1], 0.f), S - 1.f);
    float sp  = log1pf(__expf(params[b * 3 + 2] + 2.0f));
    float sig = (sp + 0.01f) * S;

    float i0[10], i1[10];
    float fl0 = floorf(mean0), ce0 = ceilf(mean0);
    float fl1 = floorf(mean1), ce1 = ceilf(mean1);
    i0[0] = fl0; i1[0] = fl1;  i0[1] = fl0; i1[1] = ce1;
    i0[2] = ce0; i1[2] = fl1;  i0[3] = ce0; i1[3] = ce1;
    for (int g = 0; g < 2; ++g) {
      i0[4 + g] = floorf(u_glob[b * 4 + g * 2 + 0] * (1.f - E) * S);
      i1[4 + g] = floorf(u_glob[b * 4 + g * 2 + 1] * (1.f - E) * S);
    }
    const float rr = 6.0f;
    float mr0 = rintf(mean0), mr1 = rintf(mean1);
    float lo0 = (mr0 - 3.f < 0.f) ? 0.f : mr0 - 3.f;  if (mr0 + 3.f > S) lo0 = S - rr;
    float lo1 = (mr1 - 3.f < 0.f) ? 0.f : mr1 - 3.f;  if (mr1 + 3.f > S) lo1 = S - rr;
    for (int j = 0; j < 4; ++j) {
      i0[8 + j] = floorf(u_loc[b * 8 + j * 2 + 0] * (1.f - E) * rr + lo0);
      i1[8 + j] = floorf(u_loc[b * 8 + j * 2 + 1] * (1.f - E) * rr + lo1);
    }
    int flat[10];
    for (int v = 0; v < 10; ++v) flat[v] = ((int)i0[v]) * 32 + (int)i1[v];
    const float inv = rsqrtf(E + sig);
    float props[10], sum = 0.f;
    for (int v = 0; v < 10; ++v) {
      bool dup = false;
      for (int u = 0; u < v; ++u) dup = dup || (flat[v] == flat[u]);
      float z0 = (i0[v] - mean0) * inv, z1 = (i1[v] - mean1) * inv;
      float pr = dup ? 0.f : __expf(-0.5f * (z0 * z0 + z1 * z1));
      props[v] = pr; sum += pr;
    }
    for (int v = 0; v < 10; ++v) { s_w[v] = props[v] / sum; s_idx[v] = flat[v]; }
  }
  __syncthreads();

  const int e = threadIdx.x;
  float acc = 0.f;
  #pragma unroll
  for (int v = 0; v < 10; ++v)
    acc += s_w[v] * latent[(size_t)s_idx[v] * 128 + e];
  code[(size_t)b * 128 + e] = (_Float16)acc;
}

// ---------------------------------------------------------------------------
extern "C" void kernel_launch(void* const* d_in, const int* in_sizes, int n_in,
                              void* d_out, int out_size, void* d_ws, size_t ws_size,
                              hipStream_t stream)
{
  (void)in_sizes; (void)n_in; (void)out_size; (void)ws_size;

  const float* x      = (const float*)d_in[0];
  const float* latent = (const float*)d_in[1];
  const float* c0a_w = (const float*)d_in[2];  const float* c0a_b = (const float*)d_in[3];
  const float* c0b_w = (const float*)d_in[4];  const float* c0b_b = (const float*)d_in[5];
  const float* c1a_w = (const float*)d_in[6];  const float* c1a_b = (const float*)d_in[7];
  const float* c1b_w = (const float*)d_in[8];  const float* c1b_b = (const float*)d_in[9];
  const float* c2a_w = (const float*)d_in[10]; const float* c2a_b = (const float*)d_in[11];
  const float* c2b_w = (const float*)d_in[12]; const float* c2b_b = (const float*)d_in[13];
  const float* fc1_w = (const float*)d_in[14]; const float* fc1_b = (const float*)d_in[15];
  const float* fc2_w = (const float*)d_in[16]; const float* fc2_b = (const float*)d_in[17];
  const float* dfc_w = (const float*)d_in[18]; const float* dfc_b = (const float*)d_in[19];
  const float* t3a_w = (const float*)d_in[20]; const float* t3a_b = (const float*)d_in[21];
  const float* t3b_w = (const float*)d_in[22]; const float* t3b_b = (const float*)d_in[23];
  const float* t2a_w = (const float*)d_in[24]; const float* t2a_b = (const float*)d_in[25];
  const float* t2b_w = (const float*)d_in[26]; const float* t2b_b = (const float*)d_in[27];
  const float* t1a_w = (const float*)d_in[28]; const float* t1a_b = (const float*)d_in[29];
  const float* t1b_w = (const float*)d_in[30]; const float* t1b_b = (const float*)d_in[31];
  const float* t0_w  = (const float*)d_in[32]; const float* t0_b  = (const float*)d_in[33];
  const float* u_glob = (const float*)d_in[34];
  const float* u_loc  = (const float*)d_in[35];

  float* out = (float*)d_out;

  _Float16* ws = (_Float16*)d_ws;
  size_t off = 0;
  auto alloc = [&](size_t nelem) {
    _Float16* pp = ws + off; off += (nelem + 7) & ~(size_t)7; return pp;
  };
  const size_t BUFN = 8388608;
  _Float16* buf0 = alloc(BUFN);
  _Float16* buf1 = alloc(BUFN);
  _Float16* xh   = alloc((size_t)128 * 4096 * 3);
  _Float16* fc1T = alloc((size_t)1024 * 4096);
  _Float16* dfcT = alloc((size_t)4096 * 128);
  _Float16* w0a = alloc(9*16*32);  _Float16* w0b = alloc(9*16*32);
  _Float16* w1a = alloc(9*32*32);  _Float16* w1b = alloc(9*32*32);
  _Float16* w2a = alloc(9*64*32);  _Float16* w2b = alloc(9*64*64);
  _Float16* d3a = alloc(9*64*64);  _Float16* d3b = alloc(9*32*64);
  _Float16* d2a = alloc(9*32*32);  _Float16* d2b = alloc(9*16*32);
  _Float16* d1a = alloc(9*16*32);  _Float16* d1b = alloc(9*16*32);
  _Float16* d0  = alloc(9*16*32);
  _Float16* codeh = alloc(128 * 128);
  float* prm = (float*)(ws + off);

  const int B = 128;
  const dim3 cblk(128);

  pack_x<<<(128*4096*3 + 255)/256, 256, 0, stream>>>(x, xh);
  pack_conv_w<<<(9*16*32 + 255)/256, 256, 0, stream>>>(c0a_w, w0a, 16, 3, 16, 32);
  pack_conv_w<<<(9*16*32 + 255)/256, 256, 0, stream>>>(c0b_w, w0b, 16, 16, 16, 32);
  pack_conv_w<<<(9*32*32 + 255)/256, 256, 0, stream>>>(c1a_w, w1a, 32, 16, 32, 32);
  pack_conv_w<<<(9*32*32 + 255)/256, 256, 0, stream>>>(c1b_w, w1b, 32, 32, 32, 32);
  pack_conv_w<<<(9*64*32 + 255)/256, 256, 0, stream>>>(c2a_w, w2a, 64, 32, 64, 32);
  pack_conv_w<<<(9*64*64 + 255)/256, 256, 0, stream>>>(c2b_w, w2b, 64, 64, 64, 64);
  pack_conv_w<<<(9*64*64 + 255)/256, 256, 0, stream>>>(t3a_w, d3a, 64, 64, 64, 64);
  pack_conv_w<<<(9*32*64 + 255)/256, 256, 0, stream>>>(t3b_w, d3b, 32, 64, 32, 64);
  pack_conv_w<<<(9*32*32 + 255)/256, 256, 0, stream>>>(t2a_w, d2a, 32, 32, 32, 32);
  pack_conv_w<<<(9*16*32 + 255)/256, 256, 0, stream>>>(t2b_w, d2b, 16, 32, 16, 32);
  pack_conv_w<<<(9*16*32 + 255)/256, 256, 0, stream>>>(t1a_w, d1a, 16, 16, 16, 32);
  pack_conv_w<<<(9*16*32 + 255)/256, 256, 0, stream>>>(t1b_w, d1b, 3, 16, 16, 32);
  pack_conv_w<<<(9*16*32 + 255)/256, 256, 0, stream>>>(t0_w,  d0,  3, 3, 16, 32);
  pack_fc1_w<<<((1024*4096) + 255)/256, 256, 0, stream>>>(fc1_w, fc1T);
  pack_dfc_w<<<((4096*128) + 255)/256, 256, 0, stream>>>(dfc_w, dfcT);

  // ---- encoder ----
  conv3x3_wmma<3, 16, 64, 64, 1, false, 1><<<dim3(64, 1, B), cblk, 0, stream>>>(xh,   w0a, c0a_b, buf0);
  conv3x3_wmma<16,16, 64, 64, 1, false, 1><<<dim3(64, 1, B), cblk, 0, stream>>>(buf0, w0b, c0b_b, buf1);
  maxpool2_nhwc<<<(B*32*32*16 + 255)/256, 256, 0, stream>>>(buf1, buf0, B, 64, 64, 16);

  conv3x3_wmma<16,32, 32, 32, 1, false, 2><<<dim3(16, 1, B), cblk, 0, stream>>>(buf0, w1a, c1a_b, buf1);
  conv3x3_wmma<32,32, 32, 32, 1, false, 2><<<dim3(16, 1, B), cblk, 0, stream>>>(buf1, w1b, c1b_b, buf0);
  maxpool2_nhwc<<<(B*16*16*32 + 255)/256, 256, 0, stream>>>(buf0, buf1, B, 32, 32, 32);

  conv3x3_wmma<32,64, 16, 16, 1, false, 4><<<dim3(4, 1, B), cblk, 0, stream>>>(buf1, w2a, c2a_b, buf0);
  conv3x3_wmma<64,64, 16, 16, 1, false, 4><<<dim3(4, 1, B), cblk, 0, stream>>>(buf0, w2b, c2b_b, buf1);
  maxpool2_nhwc<<<(B*8*8*64 + 255)/256, 256, 0, stream>>>(buf1, buf0, B, 16, 16, 64);
  // buf0 = (B, 8, 8, 64) NHWC = (B, 4096) rows for fc1

  gemm_wmma<1, false, 2><<<dim3(8, 32), dim3(32), 0, stream>>>(buf0, fc1T, fc1_b, buf1, 1024, 4096);
  fc2_kernel<<<3, 128, 0, stream>>>(buf1, fc2_w, fc2_b, prm);

  sample_gather_kernel<<<B, 128, 0, stream>>>(prm, u_glob, u_loc, latent, codeh);

  // ---- decoder ----
  gemm_wmma<1, true, 2><<<dim3(8, 128), dim3(32), 0, stream>>>(codeh, dfcT, dfc_b, buf0, 4096, 128);

  up2_nhwc<<<(B*16*16*64 + 255)/256, 256, 0, stream>>>(buf0, buf1, B, 8, 8, 64);
  conv3x3_wmma<64,64, 16, 16, 1, false, 4><<<dim3(4, 1, B), cblk, 0, stream>>>(buf1, d3a, t3a_b, buf0);
  conv3x3_wmma<64,32, 16, 16, 1, false, 2><<<dim3(4, 1, B), cblk, 0, stream>>>(buf0, d3b, t3b_b, buf1);

  up2_nhwc<<<(B*32*32*32 + 255)/256, 256, 0, stream>>>(buf1, buf0, B, 16, 16, 32);
  conv3x3_wmma<32,32, 32, 32, 1, false, 2><<<dim3(16, 1, B), cblk, 0, stream>>>(buf0, d2a, t2a_b, buf1);
  conv3x3_wmma<32,16, 32, 32, 1, false, 1><<<dim3(16, 1, B), cblk, 0, stream>>>(buf1, d2b, t2b_b, buf0);

  up2_nhwc<<<(B*64*64*16 + 255)/256, 256, 0, stream>>>(buf0, buf1, B, 32, 32, 16);
  conv3x3_wmma<16,16, 64, 64, 1, false, 1><<<dim3(64, 1, B), cblk, 0, stream>>>(buf1, d1a, t1a_b, buf0);
  conv3x3_wmma<16, 3, 64, 64, 1, false, 1><<<dim3(64, 1, B), cblk, 0, stream>>>(buf0, d1b, t1b_b, buf1);

  conv3x3_wmma<3, 3, 64, 64, 2, true, 1><<<dim3(64, 1, B), cblk, 0, stream>>>(buf1, d0, t0_b, out);
}
